// AttentionAggregator_57088705298647
// MI455X (gfx1250) — matrix operand order repaired and checked by reading, not compiled, not run on hardware
//
#include <hip/hip_runtime.h>
#include <hip/hip_bf16.h>
#include <stdint.h>

#define N_NODES 4096
#define DEG 16
#define DIM 64
#define HID 128
#define KPROJ (DEG * 2 * DIM) /* 2048 */

typedef __attribute__((ext_vector_type(16))) __bf16 v16bf;
typedef __attribute__((ext_vector_type(8)))  float  v8f;

union BF16Frag {
  v16bf v;
  uint16_t u[16];
  uint4 q[2];
};

static __device__ __forceinline__ uint16_t f2bf(float f) {
  union { float f; uint32_t u; } x; x.f = f;
  uint32_t r = x.u + 0x7FFFu + ((x.u >> 16) & 1u); // round-to-nearest-even
  return (uint16_t)(r >> 16);
}

static __device__ __forceinline__ v8f wmma_bf16(v16bf a, v16bf b, v8f c) {
  // D = A(16x32 bf16) * B(32x16 bf16) + C(16x16 f32)
  return __builtin_amdgcn_wmma_f32_16x16x32_bf16(false, a, false, b, (short)0, c,
                                                 false, false);
}

// ---------------------------------------------------------------------------
// K1: fp32 -> bf16 row-major convert
// ---------------------------------------------------------------------------
__global__ __launch_bounds__(256) void convert_kernel(const float* __restrict__ src,
                                                      uint16_t* __restrict__ dst,
                                                      int n) {
  int i = blockIdx.x * 256 + threadIdx.x;
  if (i < n) dst[i] = f2bf(src[i]);
}

// ---------------------------------------------------------------------------
// K2: repack W [KPROJ][HID] fp32 into WMMA B-fragment order (bf16):
//   dst[((c*8 + b)*32 + lane)*16 + h]  <-  W[(c*32 + (lane>=16)*16 + h)*HID + b*16 + (lane&15)]
// so each lane's 16 halfs per (k-chunk c, n-block b) are one contiguous 32B read.
// ---------------------------------------------------------------------------
__global__ __launch_bounds__(256) void pack_weights_kernel(const float* __restrict__ W,
                                                           uint16_t* __restrict__ dst) {
  int tid = blockIdx.x * 256 + threadIdx.x;
  if (tid >= KPROJ * HID) return;
  int h    = tid & 15;
  int lane = (tid >> 4) & 31;
  int b    = (tid >> 9) & 7;
  int c    = tid >> 12;
  int k = c * 32 + ((lane >> 4) << 4) + h;
  int n = b * 16 + (lane & 15);
  dst[tid] = f2bf(W[(size_t)k * HID + n]);
}

// ---------------------------------------------------------------------------
// K3: gather + projection.  One block = 16 nodes.  8 waves, each wave owns a
// 16-wide HID slice.  A-tile (16 x 2048 bf16, 64 KB LDS) gathered by 256
// threads (one neighbor pair each).  Output written transposed [HID][N] bf16.
// ---------------------------------------------------------------------------
__global__ __launch_bounds__(256) void proj_kernel(
    const float* __restrict__ srcA, const float* __restrict__ srcB,
    const int* __restrict__ adjA, const int* __restrict__ adjB,
    const uint16_t* __restrict__ Wp, uint16_t* __restrict__ Pt) {
  __shared__ uint16_t Atile[16][KPROJ]; // 64 KB

  const int base = blockIdx.x * 16;
  const int t = threadIdx.x;

  { // gather: thread t handles node row r, neighbor j
    const int r = t >> 4, j = t & 15;
    const int ia = adjA[(size_t)(base + r) * DEG + j];
    const int ib = adjB[(size_t)(base + r) * DEG + j];
    const float4* pa = (const float4*)(srcA + (size_t)ia * DIM);
    const float4* pb = (const float4*)(srcB + (size_t)ib * DIM);
    uint16_t* da = &Atile[r][j * 2 * DIM];
    uint16_t* db = da + DIM;
#pragma unroll
    for (int i = 0; i < DIM / 4; ++i) {
      float4 v = pa[i];
      da[i * 4 + 0] = f2bf(v.x); da[i * 4 + 1] = f2bf(v.y);
      da[i * 4 + 2] = f2bf(v.z); da[i * 4 + 3] = f2bf(v.w);
    }
#pragma unroll
    for (int i = 0; i < DIM / 4; ++i) {
      float4 v = pb[i];
      db[i * 4 + 0] = f2bf(v.x); db[i * 4 + 1] = f2bf(v.y);
      db[i * 4 + 2] = f2bf(v.z); db[i * 4 + 3] = f2bf(v.w);
    }
  }
  __syncthreads();

  const int wave = t >> 5, lane = t & 31;
  const int m = lane & 15, hi = lane >> 4;

  v8f C = {};
  for (int c = 0; c < KPROJ / 32; ++c) {
    BF16Frag a, b;
    // A 16x32 bf16 fragment: lane<16 holds k {0..15}, lane>=16 {16..31}
    const uint4* rowp = (const uint4*)&Atile[m][c * 32];
    a.q[0] = rowp[hi ? 2 : 0];
    a.q[1] = rowp[hi ? 3 : 1];
    // B fragment: pre-packed, contiguous 32B per lane
    const uint4* wp4 = (const uint4*)(Wp + (((size_t)c * 8 + wave) * 32 + lane) * 16);
    b.q[0] = wp4[0];
    b.q[1] = wp4[1];
    C = wmma_bf16(a.v, b.v, C);
  }

  // C layout: vgpr g -> row (g + 8*hi), col = lane&15.  Rows contiguous in g,
  // so pack 8 bf16 and store 16B into transposed Pt[n][node].
  const int n = wave * 16 + m;
  uint16_t h8[8];
#pragma unroll
  for (int g = 0; g < 8; ++g) h8[g] = f2bf(C[g]);
  uint4 packed;
  packed.x = (uint32_t)h8[0] | ((uint32_t)h8[1] << 16);
  packed.y = (uint32_t)h8[2] | ((uint32_t)h8[3] << 16);
  packed.z = (uint32_t)h8[4] | ((uint32_t)h8[5] << 16);
  packed.w = (uint32_t)h8[6] | ((uint32_t)h8[7] << 16);
  *(uint4*)(Pt + (size_t)n * N_NODES + base + 8 * hi) = packed;
}

// ---------------------------------------------------------------------------
// K4: attention.  Block = 16 query rows, 8 waves.
//   Pass 1: waves split keys, online row-max/row-sum, combine via LDS.
//   Pass 2: waves cooperatively build bf16 prob tiles [16][256] in LDS, each
//           wave accumulates its own 16-wide HID slice against Pt ([HID][N]).
// ---------------------------------------------------------------------------
__global__ __launch_bounds__(256) void attn_kernel(
    const uint16_t* __restrict__ Qb, const uint16_t* __restrict__ Kb,
    const uint16_t* __restrict__ Pt, float* __restrict__ Out) {
  __shared__ uint16_t Plds[16][256];  // 8 KB prob staging
  __shared__ float stats[8][16][2];
  __shared__ float gmv[16], gsv[16];

  const int base = blockIdx.x * 16;
  const int t = threadIdx.x;
  const int wave = t >> 5, lane = t & 31;
  const int m = lane & 15, hi = lane >> 4;
  const float scale = 0.125f; // 1/sqrt(64)

  // Q fragments (A-matrix), 2 k-chunks of 32 over DIM=64; row-major bf16 works
  BF16Frag qa[2];
#pragma unroll
  for (int c = 0; c < 2; ++c) {
    const uint4* qp = (const uint4*)(Qb + (size_t)(base + m) * DIM + c * 32);
    qa[c].q[0] = qp[hi ? 2 : 0];
    qa[c].q[1] = qp[hi ? 3 : 1];
  }

  // ---- pass 1: softmax stats ----
  float rm[8], rs[8];
#pragma unroll
  for (int g = 0; g < 8; ++g) { rm[g] = -3.0e38f; rs[g] = 0.f; }

  for (int kb = 0; kb < N_NODES / 128; ++kb) {
    const int key0 = kb * 128 + wave * 16;
    v8f S = {};
#pragma unroll
    for (int c = 0; c < 2; ++c) {
      BF16Frag kf; // B = K^T: lane n=key, k contiguous in memory
      const uint4* kp = (const uint4*)(Kb + (size_t)(key0 + m) * DIM + c * 32 + hi * 16);
      kf.q[0] = kp[0];
      kf.q[1] = kp[1];
      S = wmma_bf16(qa[c].v, kf.v, S);
    }
#pragma unroll
    for (int g = 0; g < 8; ++g) {
      float v = S[g] * scale;
      float mx = v;
      for (int d = 1; d < 16; d <<= 1) mx = fmaxf(mx, __shfl_xor(mx, d, 32));
      float nm = fmaxf(rm[g], mx);
      float sm = __expf(v - nm);
      for (int d = 1; d < 16; d <<= 1) sm += __shfl_xor(sm, d, 32);
      rs[g] = rs[g] * __expf(rm[g] - nm) + sm;
      rm[g] = nm;
    }
  }
  if (m == 0) {
#pragma unroll
    for (int g = 0; g < 8; ++g) {
      stats[wave][g + 8 * hi][0] = rm[g];
      stats[wave][g + 8 * hi][1] = rs[g];
    }
  }
  __syncthreads();
  if (t < 16) {
    float M = -3.0e38f;
    for (int w = 0; w < 8; ++w) M = fmaxf(M, stats[w][t][0]);
    float Ss = 0.f;
    for (int w = 0; w < 8; ++w) Ss += stats[w][t][1] * __expf(stats[w][t][0] - M);
    gmv[t] = M;
    gsv[t] = Ss;
  }
  __syncthreads();
  float lgm[8];
#pragma unroll
  for (int g = 0; g < 8; ++g) lgm[g] = gmv[g + 8 * hi];

  // ---- pass 2: O = P @ Pt-slice ----
  v8f O = {};
  for (int kb = 0; kb < N_NODES / 256; ++kb) {
    // each wave produces probs for its 32 keys (2 tiles of 16)
#pragma unroll
    for (int tt = 0; tt < 2; ++tt) {
      const int key0 = kb * 256 + wave * 32 + tt * 16;
      v8f S = {};
#pragma unroll
      for (int c = 0; c < 2; ++c) {
        BF16Frag kf;
        const uint4* kp = (const uint4*)(Kb + (size_t)(key0 + m) * DIM + c * 32 + hi * 16);
        kf.q[0] = kp[0];
        kf.q[1] = kp[1];
        S = wmma_bf16(qa[c].v, kf.v, S);
      }
#pragma unroll
      for (int g = 0; g < 8; ++g)
        Plds[g + 8 * hi][wave * 32 + tt * 16 + m] = f2bf(__expf(S[g] * scale - lgm[g]));
    }
    __syncthreads();
    // consume shared prob tile: 8 WMMAs of K=32 against this wave's HID slice
#pragma unroll
    for (int c = 0; c < 8; ++c) {
      BF16Frag a, b;
      const uint4* rowp = (const uint4*)&Plds[m][c * 32];
      a.q[0] = rowp[hi ? 2 : 0];
      a.q[1] = rowp[hi ? 3 : 1];
      const int hcol = wave * 16 + m;
      const uint4* pp =
          (const uint4*)(Pt + (size_t)hcol * N_NODES + kb * 256 + c * 32 + hi * 16);
      b.q[0] = pp[0];
      b.q[1] = pp[1];
      O = wmma_bf16(a.v, b.v, O);
    }
    __syncthreads();
  }

  // epilogue: divide by softmax denom, relu, store fp32
#pragma unroll
  for (int g = 0; g < 8; ++g) {
    float v = O[g] / gsv[g + 8 * hi];
    v = v > 0.f ? v : 0.f;
    Out[(size_t)(base + g + 8 * hi) * HID + wave * 16 + m] = v;
  }
}

// ---------------------------------------------------------------------------
extern "C" void kernel_launch(void* const* d_in, const int* in_sizes, int n_in,
                              void* d_out, int out_size, void* d_ws, size_t ws_size,
                              hipStream_t stream) {
  (void)in_sizes; (void)n_in; (void)out_size; (void)ws_size;
  const float* review = (const float*)d_in[0];
  const float* uvec   = (const float*)d_in[1];
  const float* ivec   = (const float*)d_in[2];
  const float* Wu     = (const float*)d_in[3];
  const float* Wi     = (const float*)d_in[4];
  const int* adj0 = (const int*)d_in[5];
  const int* adj1 = (const int*)d_in[6];
  const int* adj2 = (const int*)d_in[7];
  const int* adj3 = (const int*)d_in[8];
  float* out = (float*)d_out;

  char* ws = (char*)d_ws;
  uint16_t* Qu  = (uint16_t*)(ws);                   // 512 KB: user vecs bf16
  uint16_t* Qi  = (uint16_t*)(ws + (512u << 10));    // 512 KB: item vecs bf16
  uint16_t* Wup = (uint16_t*)(ws + (1024u << 10));   // 512 KB: packed user W
  uint16_t* Wip = (uint16_t*)(ws + (1536u << 10));   // 512 KB: packed item W
  uint16_t* PtU = (uint16_t*)(ws + (2048u << 10));   // 1 MB: user P transposed
  uint16_t* PtI = (uint16_t*)(ws + (3072u << 10));   // 1 MB: item P transposed

  convert_kernel<<<(N_NODES * DIM) / 256, 256, 0, stream>>>(uvec, Qu, N_NODES * DIM);
  convert_kernel<<<(N_NODES * DIM) / 256, 256, 0, stream>>>(ivec, Qi, N_NODES * DIM);
  pack_weights_kernel<<<(KPROJ * HID) / 256, 256, 0, stream>>>(Wu, Wup);
  pack_weights_kernel<<<(KPROJ * HID) / 256, 256, 0, stream>>>(Wi, Wip);

  proj_kernel<<<N_NODES / 16, 256, 0, stream>>>(review, ivec, adj0, adj1, Wup, PtU);
  proj_kernel<<<N_NODES / 16, 256, 0, stream>>>(review, uvec, adj2, adj3, Wip, PtI);

  attn_kernel<<<N_NODES / 16, 256, 0, stream>>>(Qu, Qu, PtU, out);
  attn_kernel<<<N_NODES / 16, 256, 0, stream>>>(Qi, Qi, PtI, out + (size_t)N_NODES * HID);
}
